// FreqAttnPlus_10866267259305
// MI455X (gfx1250) — matrix operand order repaired and checked by reading, not compiled, not run on hardware
//
#include <hip/hip_runtime.h>
#include <hip/hip_bf16.h>

// ---------- types for WMMA ----------
typedef __attribute__((ext_vector_type(16))) __bf16 v16bf;
typedef __attribute__((ext_vector_type(8)))  float  v8f;
typedef __attribute__((ext_vector_type(4)))  unsigned int u32x4;

union FragU { u32x4 q[2]; v16bf v; };

__device__ __forceinline__ unsigned short f2bf(float f) {
    unsigned int u = __float_as_uint(f);
    u += 0x7FFFu + ((u >> 16) & 1u);
    return (unsigned short)(u >> 16);
}

#define HW    4096
#define GSTR  40          // LDS row stride in ushorts (80B: 16B-aligned, bank-spread)

// ---------- async global->LDS copy path (gfx1250), guarded ----------
#if defined(__has_builtin)
#  if __has_builtin(__builtin_amdgcn_global_load_async_to_lds_b128)
#    define ASYNC_G2L 1
#  endif
#  if __has_builtin(__builtin_amdgcn_s_wait_asynccnt)
#    define HAVE_WAIT_ASYNC 1
#  endif
#endif

// builtin signature (from compiler diagnostic): param1 = AS1 ptr to 16B int
// vector (non-const), param2 = AS3 ptr, then imm offset / imm cpol.
typedef int v4i_vs __attribute__((__vector_size__(16)));
typedef __attribute__((address_space(1))) v4i_vs as1_v4i;
typedef __attribute__((address_space(3))) v4i_vs as3_v4i;

__device__ __forceinline__ as1_v4i* as_gbl(const void* p) {
    return (as1_v4i*)(unsigned long long)p;           // strip const + AS via int
}
__device__ __forceinline__ as3_v4i* as_lds(void* p) {
    // generic LDS pointer: low 32 bits are the LDS byte offset (ISA 10.2)
    return (as3_v4i*)(unsigned int)(unsigned long long)p;
}

__device__ __forceinline__ void g2l_b128(float* ldst, const float* gsrc) {
#if ASYNC_G2L
    __builtin_amdgcn_global_load_async_to_lds_b128(as_gbl(gsrc), as_lds(ldst), 0, 0);
#else
    *(float4*)ldst = *(const float4*)gsrc;
#endif
}
// wait for this wave's async copies, then workgroup barrier
__device__ __forceinline__ void g2l_fence() {
#if ASYNC_G2L
#  if HAVE_WAIT_ASYNC
    __builtin_amdgcn_s_wait_asynccnt(0);
#  else
    asm volatile("s_wait_asynccnt 0x0" ::: "memory");
#  endif
#endif
    __syncthreads();
}

// =====================================================================
// Tiled bf16 WMMA GEMM:  out[b,m,n] = sum_k W[m,k] * src(b,k,n) (+bias)(+resid)
// Block tile 128(M) x 64(N), 8 waves, wave = 16(M) x 64(N) strip.
// K, Mtot compile-time so staging loads use constant instruction offsets.
// SPLIT: k<256 reads src0, else src1 (both [B,256,HW]); else src0 is [B,K,HW].
// =====================================================================
template<int K, int Mtot, bool SPLIT, bool RESID>
__global__ __launch_bounds__(256)
void k_gemm(const float* __restrict__ W,
            const float* __restrict__ src0, const float* __restrict__ src1,
            const float* __restrict__ bias, const float* __restrict__ resid,
            float* __restrict__ out)
{
    __shared__ __attribute__((aligned(16))) unsigned short sA[128 * GSTR];
    __shared__ __attribute__((aligned(16))) unsigned short sB[64 * GSTR];

    const int mblk = blockIdx.x >> 6;          // HW/64 = 64 n-blocks
    const int nblk = blockIdx.x & 63;
    const int b    = blockIdx.y;
    const int m0   = mblk * 128, n0 = nblk * 64;
    const int t    = threadIdx.x;
    const int wave = t >> 5, lane = t & 31;
    const int r16  = lane & 15, kh = lane >> 4;
    const int tr5 = t >> 5, tc5 = t & 31;      // A staging coords
    const int tr6 = t >> 6, tc6 = t & 63;      // B staging coords

    v8f acc[4];
    #pragma unroll
    for (int i = 0; i < 4; ++i) acc[i] = v8f{0.f,0.f,0.f,0.f,0.f,0.f,0.f,0.f};

    for (int kb = 0; kb < K; kb += 32) {
        // ---- stage A tile: rows m0..m0+127, cols kb..kb+31 ([M][K] row-major) ----
        const float* ap = W + (size_t)(m0 + tr5) * K + kb + tc5;
        #pragma unroll
        for (int i = 0; i < 16; ++i)
            sA[(tr5 + 8 * i) * GSTR + tc5] = f2bf(ap[(size_t)(8 * i) * K]);

        // ---- stage B tile transposed: sB[n][k]; source choice uniform per slice ----
        const float* srcx;
        size_t chan;
        if (SPLIT) {
            const bool lo = kb < 256;
            srcx = lo ? src0 : src1;
            chan = (size_t)b * 256 + (lo ? kb : kb - 256);
        } else {
            srcx = src0;
            chan = (size_t)b * K + kb;
        }
        const float* bp = srcx + (chan + tr6) * HW + n0 + tc6;
        #pragma unroll
        for (int i = 0; i < 8; ++i)
            sB[tc6 * GSTR + tr6 + 4 * i] = f2bf(bp[(size_t)(4 * i) * HW]);
        if (kb + 32 < K && (!SPLIT || kb + 32 != 256))
            __builtin_prefetch(bp + (size_t)32 * HW, 0, 1);
        __syncthreads();

        // ---- A fragment (16-bit A layout: K halves 8kh.., 16+8kh..) ----
        const unsigned short* arow = sA + (wave * 16 + r16) * GSTR;
        FragU af;
        af.q[0] = *(const u32x4*)(arow + 8 * kh);
        af.q[1] = *(const u32x4*)(arow + 16 + 8 * kh);

        #pragma unroll
        for (int nt = 0; nt < 4; ++nt) {
            // ---- B fragment (16-bit layout: K halves 8kh.., 16+8kh..) ----
            const unsigned short* brow = sB + (nt * 16 + r16) * GSTR;
            FragU bfr;
            bfr.q[0] = *(const u32x4*)(brow + 8 * kh);
            bfr.q[1] = *(const u32x4*)(brow + 16 + 8 * kh);
            acc[nt] = __builtin_amdgcn_wmma_f32_16x16x32_bf16(
                false, af.v, false, bfr.v, (short)0, acc[nt], false, false);
        }
        __syncthreads();
    }

    // ---- writeback: C/D layout: row m = mhalf*8+j, col n = lane&15 ----
    #pragma unroll
    for (int nt = 0; nt < 4; ++nt) {
        #pragma unroll
        for (int j = 0; j < 8; ++j) {
            int m = m0 + wave * 16 + kh * 8 + j;
            int n = n0 + nt * 16 + r16;
            float v = acc[nt][j] + bias[m];
            size_t o = ((size_t)b * Mtot + m) * HW + n;
            if (RESID) v += resid[o];
            out[o] = v;
        }
    }
}

// =====================================================================
// Depthwise 3x3 + BN(eval) + ReLU, fused with SE average-pool partial.
// =====================================================================
__global__ __launch_bounds__(256)
void k_dw(const float* __restrict__ xr0, const float* __restrict__ dww,
          const float* __restrict__ dwb, const float* __restrict__ gamma,
          const float* __restrict__ beta, const float* __restrict__ mean,
          const float* __restrict__ var, float* __restrict__ xr,
          float* __restrict__ pool)
{
    __shared__ float sp[64 * 66];
    __shared__ float part[256];
    const int bc = blockIdx.x;          // b*256 + c
    const int c  = bc & 255;
    const int t  = threadIdx.x;
    const float* in = xr0 + (size_t)bc * HW;

    #pragma unroll
    for (int i = 0; i < 16; ++i) {
        int idx = i * 256 + t;
        sp[(idx >> 6) * 66 + (idx & 63)] = in[idx];
    }
    __syncthreads();

    float w[9];
    #pragma unroll
    for (int i = 0; i < 9; ++i) w[i] = dww[c * 9 + i];
    const float bnb = dwb[c];
    const float inv = gamma[c] * rsqrtf(var[c] + 1e-5f);
    const float mn = mean[c], bt = beta[c];

    float lsum = 0.f;
    #pragma unroll 4
    for (int i = 0; i < 16; ++i) {
        int idx = i * 256 + t;
        int y = idx >> 6, x = idx & 63;
        float s = 0.f;
        #pragma unroll
        for (int dy = -1; dy <= 1; ++dy)
            #pragma unroll
            for (int dx = -1; dx <= 1; ++dx) {
                int yy = y + dy, xx = x + dx;
                float v = (yy >= 0 && yy < 64 && xx >= 0 && xx < 64)
                              ? sp[yy * 66 + xx] : 0.f;
                s += v * w[(dy + 1) * 3 + (dx + 1)];
            }
        s = (s + bnb - mn) * inv + bt;
        s = fmaxf(s, 0.f);
        xr[(size_t)bc * HW + idx] = s;
        lsum += s;
    }
    part[t] = lsum;
    __syncthreads();
    for (int s2 = 128; s2 > 0; s2 >>= 1) {
        if (t < s2) part[t] += part[t + s2];
        __syncthreads();
    }
    if (t == 0) pool[bc] = part[0] * (1.f / 4096.f);
}

// =====================================================================
// In-LDS 64x64 radix-2 FFT2 machinery (256 threads per plane).
// Stride 65 floats: conflict-free for both row and column butterflies.
// =====================================================================
__device__ __forceinline__ int brev6(int x) { return (int)(__brev((unsigned)x) >> 26); }

__device__ void bitrev_cols(float* re, float* im, int t) {
    #pragma unroll 4
    for (int i = 0; i < 16; ++i) {
        int idx = i * 256 + t;
        int r = idx >> 6, c = idx & 63, bc = brev6(c);
        if (c < bc) {
            int a = r * 65 + c, b2 = r * 65 + bc;
            float tr = re[a]; re[a] = re[b2]; re[b2] = tr;
            float ti = im[a]; im[a] = im[b2]; im[b2] = ti;
        }
    }
}
__device__ void bitrev_rows(float* re, float* im, int t) {
    #pragma unroll 4
    for (int i = 0; i < 16; ++i) {
        int idx = i * 256 + t;
        int r = idx >> 6, c = idx & 63, br = brev6(r);
        if (r < br) {
            int a = r * 65 + c, b2 = br * 65 + c;
            float tr = re[a]; re[a] = re[b2]; re[b2] = tr;
            float ti = im[a]; im[a] = im[b2]; im[b2] = ti;
        }
    }
}
__device__ void fft64_pass(float* re, float* im, int t, float dirSign, bool cols) {
    for (int s = 0; s < 6; ++s) {
        int half = 1 << s;
        __syncthreads();
        #pragma unroll 2
        for (int j = 0; j < 8; ++j) {
            int idx = j * 256 + t;
            int line = idx >> 5, bidx = idx & 31;
            int pos = bidx & (half - 1);
            int grp = bidx >> s;
            int e0 = grp * (half << 1) + pos;
            int e1 = e0 + half;
            int i0 = cols ? (e0 * 65 + line) : (line * 65 + e0);
            int i1 = cols ? (e1 * 65 + line) : (line * 65 + e1);
            float ang = dirSign * 6.28318530717958647692f * (float)pos / (float)(half << 1);
            float sn, cs;
            __sincosf(ang, &sn, &cs);
            float xr = re[i1], xi = im[i1];
            float tr = cs * xr - sn * xi;
            float ti = cs * xi + sn * xr;
            float ar = re[i0], ai = im[i0];
            re[i1] = ar - tr; im[i1] = ai - ti;
            re[i0] = ar + tr; im[i0] = ai + ti;
        }
    }
}
__device__ void fft2_inplace(float* re, float* im, int t, float dirSign) {
    __syncthreads();
    bitrev_cols(re, im, t);
    fft64_pass(re, im, t, dirSign, false);
    __syncthreads();
    bitrev_rows(re, im, t);
    fft64_pass(re, im, t, dirSign, true);
    __syncthreads();
}

// ---- forward ortho FFT2, fused with per-row L2 norm reduction ----
__global__ __launch_bounds__(256)
void k_fftfwd(const float* __restrict__ xr, float* __restrict__ Fre,
              float* __restrict__ Fim, float* __restrict__ rnorm)
{
    __shared__ float re[64 * 65], im[64 * 65];
    __shared__ float part[256];
    const int bc = blockIdx.x;
    const int t  = threadIdx.x;
    const float* in = xr + (size_t)bc * HW;
    #pragma unroll
    for (int i = 0; i < 4; ++i) {
        int idx4 = i * 256 + t;                // float4 granule
        float4 v = ((const float4*)in)[idx4];
        int a = ((idx4 * 4) >> 6) * 65 + ((idx4 * 4) & 63);
        re[a] = v.x; re[a + 1] = v.y; re[a + 2] = v.z; re[a + 3] = v.w;
        im[a] = 0.f; im[a + 1] = 0.f; im[a + 2] = 0.f; im[a + 3] = 0.f;
    }
    fft2_inplace(re, im, t, -1.f);
    float lsum = 0.f;
    #pragma unroll 4
    for (int i = 0; i < 16; ++i) {
        int idx = i * 256 + t;
        int a = (idx >> 6) * 65 + (idx & 63);
        float r = re[a] * (1.f / 64.f), m = im[a] * (1.f / 64.f);
        Fre[(size_t)bc * HW + idx] = r;
        Fim[(size_t)bc * HW + idx] = m;
        lsum += r * r + m * m;
    }
    part[t] = lsum;
    __syncthreads();
    for (int s = 128; s > 0; s >>= 1) {
        if (t < s) part[t] += part[t + s];
        __syncthreads();
    }
    if (t == 0) rnorm[bc] = fmaxf(sqrtf(part[0]), 1e-12f);
}

// ---- SE gate MLP ----
__global__ __launch_bounds__(256)
void k_gate(const float* __restrict__ pool, const float* __restrict__ w1,
            const float* __restrict__ b1, const float* __restrict__ w2,
            const float* __restrict__ b2, float* __restrict__ g)
{
    __shared__ float spv[256], sh[32];
    const int b = blockIdx.x, t = threadIdx.x;
    spv[t] = pool[b * 256 + t];
    __syncthreads();
    if (t < 32) {
        float s = b1[t];
        for (int c = 0; c < 256; ++c) s += w1[t * 256 + c] * spv[c];
        sh[t] = fmaxf(s, 0.f);
    }
    __syncthreads();
    float s = b2[t];
    #pragma unroll
    for (int j = 0; j < 32; ++j) s += w2[t * 32 + j] * sh[j];
    g[b * 256 + t] = 1.f / (1.f + __expf(-s));
}

// ---- attention Gram matrix (real part of complex, no conjugate) ----
// LDS tiles staged with async global->LDS b128 copies (stride 68 = 17x16B).
__global__ __launch_bounds__(256)
void k_attn(const float* __restrict__ Fre, const float* __restrict__ Fim,
            const float* __restrict__ rnorm, const float* __restrict__ temp,
            float* __restrict__ attnb)
{
    __shared__ __attribute__((aligned(16))) float qr[32 * 68];
    __shared__ __attribute__((aligned(16))) float qi[32 * 68];
    const int bh = blockIdx.x;               // b*8 + h
    const int b = bh >> 3, h = bh & 7;
    const int t = threadIdx.x;
    const size_t base = ((size_t)b * 256 + h * 32) * HW;
    float acc[4] = {0.f, 0.f, 0.f, 0.f};

    for (int ch = 0; ch < 64; ++ch) {
        __syncthreads();
        #pragma unroll
        for (int i = 0; i < 2; ++i) {
            int idx4 = i * 256 + t;          // 512 float4s per array
            int r = idx4 >> 4, c4 = idx4 & 15;
            g2l_b128(qr + r * 68 + c4 * 4,
                     Fre + base + (size_t)r * HW + ch * 64 + c4 * 4);
            g2l_b128(qi + r * 68 + c4 * 4,
                     Fim + base + (size_t)r * HW + ch * 64 + c4 * 4);
        }
        g2l_fence();
        #pragma unroll
        for (int i = 0; i < 4; ++i) {
            int p = i * 256 + t;
            int c = p >> 5, d = p & 31;
            const float* r0 = qr + c * 68;
            const float* i0 = qi + c * 68;
            const float* r1 = qr + d * 68;
            const float* i1 = qi + d * 68;
            float s = acc[i];
            #pragma unroll 8
            for (int n = 0; n < 64; ++n)
                s += r0[n] * r1[n] - i0[n] * i1[n];
            acc[i] = s;
        }
    }
    const float tv = temp[h];
    #pragma unroll
    for (int i = 0; i < 4; ++i) {
        int p = i * 256 + t;
        int c = p >> 5, d = p & 31;
        float nc = rnorm[b * 256 + h * 32 + c];
        float nd = rnorm[b * 256 + h * 32 + d];
        attnb[(size_t)bh * 1024 + p] = acc[i] / (nc * nd) * tv;
    }
}

// ---- row softmax over 32x32 attention ----
__global__ void k_softmax(float* __restrict__ attnb)
{
    const int bh = blockIdx.x;
    const int c = threadIdx.x;
    if (c < 32) {
        float* row = attnb + (size_t)bh * 1024 + c * 32;
        float mx = row[0];
        #pragma unroll
        for (int d = 1; d < 32; ++d) mx = fmaxf(mx, row[d]);
        float e[32], s = 0.f;
        #pragma unroll
        for (int d = 0; d < 32; ++d) { e[d] = __expf(row[d] - mx); s += e[d]; }
        float invs = 1.f / s;
        #pragma unroll
        for (int d = 0; d < 32; ++d) row[d] = e[d] * invs;
    }
}

// ---- out = attn(32x32) x F.real(32 x 4096), async-staged LDS tiles ----
__global__ __launch_bounds__(256)
void k_outgemm(const float* __restrict__ attnb, const float* __restrict__ Fre,
               float* __restrict__ outs)
{
    __shared__ __attribute__((aligned(16))) float sq[32 * 132]; // 33x16B rows
    __shared__ float sat[32 * 32];
    const int bh = blockIdx.x;
    const int b = bh >> 3, h = bh & 7;
    const int t = threadIdx.x;
    const size_t base = ((size_t)b * 256 + h * 32) * HW;
    #pragma unroll
    for (int i = 0; i < 4; ++i) sat[i * 256 + t] = attnb[(size_t)bh * 1024 + i * 256 + t];

    for (int nt = 0; nt < 32; ++nt) {
        __syncthreads();
        #pragma unroll
        for (int i = 0; i < 4; ++i) {
            int idx4 = i * 256 + t;          // 1024 float4s in tile
            int r = idx4 >> 5, c4 = idx4 & 31;
            g2l_b128(sq + r * 132 + c4 * 4,
                     Fre + base + (size_t)r * HW + nt * 128 + c4 * 4);
        }
        g2l_fence();
        #pragma unroll
        for (int i = 0; i < 16; ++i) {
            int idx = i * 256 + t;
            int c = idx >> 7, n = idx & 127;
            const float* arow = sat + c * 32;
            float s = 0.f;
            #pragma unroll 8
            for (int d = 0; d < 32; ++d) s += arow[d] * sq[d * 132 + n];
            outs[base + (size_t)c * HW + nt * 128 + n] = s;
        }
    }
}

// ---- ortho inverse FFT2 of real input, in place per plane ----
__global__ __launch_bounds__(256)
void k_ifft_out(float* __restrict__ buf)
{
    __shared__ float re[64 * 65], im[64 * 65];
    const int bc = blockIdx.x;
    const int t  = threadIdx.x;
    float* p = buf + (size_t)bc * HW;
    #pragma unroll
    for (int i = 0; i < 4; ++i) {
        int idx4 = i * 256 + t;
        float4 v = ((const float4*)p)[idx4];
        int a = ((idx4 * 4) >> 6) * 65 + ((idx4 * 4) & 63);
        re[a] = v.x; re[a + 1] = v.y; re[a + 2] = v.z; re[a + 3] = v.w;
        im[a] = 0.f; im[a + 1] = 0.f; im[a + 2] = 0.f; im[a + 3] = 0.f;
    }
    fft2_inplace(re, im, t, +1.f);
    #pragma unroll 4
    for (int i = 0; i < 16; ++i) {
        int idx = i * 256 + t;
        int a = (idx >> 6) * 65 + (idx & 63);
        p[idx] = re[a] * (1.f / 64.f);
    }
}

// ---- fwm = ifft2_unnorm( F_ortho * fft2_unnorm(g*xr) ) / HW, fully fused ----
__global__ __launch_bounds__(256)
void k_fwm(const float* __restrict__ xr, const float* __restrict__ g,
           const float* __restrict__ Fre, const float* __restrict__ Fim,
           float* __restrict__ fwm)
{
    __shared__ float re[64 * 65], im[64 * 65];
    const int bc = blockIdx.x;
    const int t  = threadIdx.x;
    const float gv = g[bc];
    const float* in = xr + (size_t)bc * HW;
    #pragma unroll
    for (int i = 0; i < 4; ++i) {
        int idx4 = i * 256 + t;
        float4 v = ((const float4*)in)[idx4];
        int a = ((idx4 * 4) >> 6) * 65 + ((idx4 * 4) & 63);
        re[a] = v.x * gv; re[a + 1] = v.y * gv; re[a + 2] = v.z * gv; re[a + 3] = v.w * gv;
        im[a] = 0.f; im[a + 1] = 0.f; im[a + 2] = 0.f; im[a + 3] = 0.f;
    }
    fft2_inplace(re, im, t, -1.f);
    // frequency-domain complex multiply by F
    #pragma unroll 4
    for (int i = 0; i < 16; ++i) {
        int idx = i * 256 + t;
        int a = (idx >> 6) * 65 + (idx & 63);
        float ar = re[a], ai = im[a];
        float br = Fre[(size_t)bc * HW + idx];
        float bi = Fim[(size_t)bc * HW + idx];
        re[a] = ar * br - ai * bi;
        im[a] = ar * bi + ai * br;
    }
    fft2_inplace(re, im, t, +1.f);
    #pragma unroll 4
    for (int i = 0; i < 16; ++i) {
        int idx = i * 256 + t;
        int a = (idx >> 6) * 65 + (idx & 63);
        fwm[(size_t)bc * HW + idx] = re[a] * (1.f / 4096.f);
    }
}

// =====================================================================
extern "C" void kernel_launch(void* const* d_in, const int* in_sizes, int n_in,
                              void* d_out, int out_size, void* d_ws, size_t ws_size,
                              hipStream_t stream)
{
    const float* x        = (const float*)d_in[0];
    const float* reduce_w = (const float*)d_in[1];
    const float* reduce_b = (const float*)d_in[2];
    const float* dw_w     = (const float*)d_in[3];
    const float* dw_b     = (const float*)d_in[4];
    const float* bn_gamma = (const float*)d_in[5];
    const float* bn_beta  = (const float*)d_in[6];
    const float* bn_mean  = (const float*)d_in[7];
    const float* bn_var   = (const float*)d_in[8];
    const float* gate_w1  = (const float*)d_in[9];
    const float* gate_b1  = (const float*)d_in[10];
    const float* gate_w2  = (const float*)d_in[11];
    const float* gate_b2  = (const float*)d_in[12];
    const float* temp     = (const float*)d_in[13];
    const float* post_w   = (const float*)d_in[14];
    const float* post_b   = (const float*)d_in[15];
    float* y = (float*)d_out;

    float* ws = (float*)d_ws;
    const size_t PL = (size_t)16 * 256 * HW;     // one [B,256,HW] plane set
    float* bufA  = ws;                            // xr0 -> outs -> out_spatial
    float* bufB  = ws + PL;                       // xr
    float* bufC  = ws + 2 * PL;                   // F.re
    float* bufD  = ws + 3 * PL;                   // F.im
    float* bufE  = ws + 4 * PL;                   // fwm
    float* pool  = ws + 5 * PL;                   // [16,256]
    float* gbuf  = pool + 4096;                   // [16,256]
    float* rnorm = gbuf + 4096;                   // [16,256]
    float* attnb = rnorm + 4096;                  // [16,8,32,32]

    dim3 blk(256);

    // 1) reduce 1x1 conv (WMMA bf16 GEMM): x[512,HW] -> xr0[256,HW] per batch
    k_gemm<512, 256, false, false><<<dim3(128, 16), blk, 0, stream>>>(
        reduce_w, x, nullptr, reduce_b, nullptr, bufA);

    // 2) depthwise 3x3 + BN + ReLU (+pool partial)
    k_dw<<<4096, blk, 0, stream>>>(bufA, dw_w, dw_b, bn_gamma, bn_beta,
                                   bn_mean, bn_var, bufB, pool);

    // 3) forward ortho FFT2 (+row norms)
    k_fftfwd<<<4096, blk, 0, stream>>>(bufB, bufC, bufD, rnorm);

    // 4) SE gate MLP
    k_gate<<<16, blk, 0, stream>>>(pool, gate_w1, gate_b1, gate_w2, gate_b2, gbuf);

    // 5) attention Gram + temperature
    k_attn<<<128, blk, 0, stream>>>(bufC, bufD, rnorm, temp, attnb);

    // 6) softmax
    k_softmax<<<128, 32, 0, stream>>>(attnb);

    // 7) out = attn x F.real  -> bufA (xr0 dead)
    k_outgemm<<<128, blk, 0, stream>>>(attnb, bufC, bufA);

    // 8) ortho inverse FFT2 of out (real part), in place
    k_ifft_out<<<4096, blk, 0, stream>>>(bufA);

    // 9) fwm = ifft2(F * fft2(g*xr)).real / HW
    k_fwm<<<4096, blk, 0, stream>>>(bufB, gbuf, bufC, bufD, bufE);

    // 10) post 1x1 conv (WMMA bf16 GEMM) over concat(out, fwm) + bias + residual
    k_gemm<512, 512, true, true><<<dim3(256, 16), blk, 0, stream>>>(
        post_w, bufA, bufE, post_b, x, y);
}
